// AttentionBlock_62380105008162
// MI455X (gfx1250) — compile-verified, matrix operand order checked by reading, not run on hardware
//
#include <hip/hip_runtime.h>
#include <hip/hip_bf16.h>

typedef __attribute__((ext_vector_type(16))) _Float16 v16h;
typedef __attribute__((ext_vector_type(8)))  float    v8f;

union FragU { v16h v; uint4 q[2]; };

// Load a 16(M)x32(K) f16 WMMA A/B fragment from row-major memory where
// row index = M (A) or N (B) and K is contiguous with leading dim `ld`.
// Per ISA 7.12.2 (16-bit A-matrix 16x32): lanes 0-15 -> K 0..7 / 16..23,
// lanes 16-31 -> K 8..15 / 24..31, i.e. koff = 8*(lane>=16).
__device__ __forceinline__ v16h load_frag16x32(const _Float16* tile, int ld) {
  const int lane = threadIdx.x & 31;
  const int row  = lane & 15;
  const int koff = (lane >> 4) << 3;   // 0 or 8
  const _Float16* p = tile + row * ld + koff;
  FragU f;
  f.q[0] = *reinterpret_cast<const uint4*>(p);        // K: koff .. koff+7
  f.q[1] = *reinterpret_cast<const uint4*>(p + 16);   // K: 16+koff .. 16+koff+7
  return f.v;
}

__device__ __forceinline__ v8f wmma32(v16h a, v16h b, v8f c) {
  return __builtin_amdgcn_wmma_f32_16x16x32_f16(false, a, false, b, (short)0, c,
                                                false, false);
}

// ---------------------------------------------------------------- constants
#define B_  16
#define C_  512
#define N_  1024
#define H_  8
#define CH_ 64
#define G_  32
#define CPG_ 16

// ------------------------------------------------------------ f32 -> f16
__global__ void cvt_f16_kernel(const float* __restrict__ a,
                               _Float16* __restrict__ o, int n) {
  int i = blockIdx.x * blockDim.x + threadIdx.x;
  if (i < n) o[i] = (_Float16)a[i];
}

// --------------------------------------------------- GroupNorm -> xnT[b][n][c]
__global__ void gn_kernel(const float* __restrict__ x,
                          const float* __restrict__ gw,
                          const float* __restrict__ gb,
                          _Float16* __restrict__ xnT) {
  const int b = blockIdx.x / G_;
  const int g = blockIdx.x % G_;
  const float* xg = x + ((size_t)b * C_ + g * CPG_) * N_;   // 16 rows x 1024
  const int t = threadIdx.x;                                 // 256 threads

  float s = 0.f, ss = 0.f;
  for (int i = t; i < CPG_ * N_; i += 256) {
    float v = xg[i];
    s += v; ss += v * v;
  }
  #pragma unroll
  for (int m = 16; m >= 1; m >>= 1) {
    s  += __shfl_xor(s,  m, 32);
    ss += __shfl_xor(ss, m, 32);
  }
  __shared__ float red[2][8];
  const int wid = t >> 5, lane = t & 31;
  if (lane == 0) { red[0][wid] = s; red[1][wid] = ss; }
  __syncthreads();
  if (t == 0) {
    float S = 0.f, SS = 0.f;
    #pragma unroll
    for (int i = 0; i < 8; ++i) { S += red[0][i]; SS += red[1][i]; }
    const float inv_n = 1.0f / (float)(CPG_ * N_);
    float mean = S * inv_n;
    float var  = SS * inv_n - mean * mean;
    red[0][0] = mean;
    red[1][0] = rsqrtf(var + 1e-5f);
  }
  __syncthreads();
  const float mean = red[0][0], rstd = red[1][0];

  // normalize + transpose through LDS: write xnT[b][n][c] (K-contiguous for WMMA B)
  __shared__ _Float16 tile[CPG_][264];   // padded vs bank conflicts
  for (int nb = 0; nb < N_; nb += 256) {
    #pragma unroll
    for (int cl = 0; cl < CPG_; ++cl) {
      const int c = g * CPG_ + cl;
      float v = (xg[cl * N_ + nb + t] - mean) * rstd * gw[c] + gb[c];
      tile[cl][t] = (_Float16)v;
    }
    __syncthreads();
    #pragma unroll
    for (int it = 0; it < 16; ++it) {
      const int nl = it * 16 + (t >> 4);
      const int cl = t & 15;
      xnT[((size_t)b * N_ + nb + nl) * C_ + g * CPG_ + cl] = tile[cl][nl];
    }
    __syncthreads();
  }
}

// ------------------------------------------------- QKV epilogue per 16x16 tile
__device__ __forceinline__ void qkv_store_tile(v8f acc, int b, int o0, int n,
                                               const float* __restrict__ bias,
                                               _Float16* __restrict__ qT,
                                               _Float16* __restrict__ kT,
                                               _Float16* __restrict__ vC) {
  const int lane = threadIdx.x & 31;
  const int off  = (lane >> 4) << 3;

  const float* bp = bias + o0 + off;
  float4 b0 = *(const float4*)bp;
  float4 b1 = *(const float4*)(bp + 4);
  float bv[8] = {b0.x, b0.y, b0.z, b0.w, b1.x, b1.y, b1.z, b1.w};
  #pragma unroll
  for (int i = 0; i < 8; ++i) acc[i] += bv[i];

  if (o0 < C_) {                                     // ---- Q (scaled)
    const int oc = o0 + off, h = oc >> 6, ch = oc & 63;
    alignas(16) _Float16 tmp[8];
    #pragma unroll
    for (int i = 0; i < 8; ++i) tmp[i] = (_Float16)(acc[i] * 0.125f);
    _Float16* dst = qT + (((size_t)b * H_ + h) * N_ + n) * CH_ + ch;
    *reinterpret_cast<uint4*>(dst) = *reinterpret_cast<const uint4*>(tmp);
  } else if (o0 < 2 * C_) {                          // ---- K
    const int oc = o0 - C_ + off, h = oc >> 6, ch = oc & 63;
    alignas(16) _Float16 tmp[8];
    #pragma unroll
    for (int i = 0; i < 8; ++i) tmp[i] = (_Float16)acc[i];
    _Float16* dst = kT + (((size_t)b * H_ + h) * N_ + n) * CH_ + ch;
    *reinterpret_cast<uint4*>(dst) = *reinterpret_cast<const uint4*>(tmp);
  } else {                                           // ---- V (channel-major)
    const int oc = o0 - 2 * C_ + off, h = oc >> 6, ch = oc & 63;
    _Float16* dst = vC + (((size_t)b * H_ + h) * CH_ + ch) * N_ + n;
    #pragma unroll
    for (int i = 0; i < 8; ++i) dst[(size_t)i * N_] = (_Float16)acc[i];
  }
}

// -------- QKV GEMM, 32x32 register-blocked: one wave per 32(M)x32(N) tile.
// q -> qT[b][h][n][64] (scaled), k -> kT[b][h][n][64], v -> vC[b][h][64][n]
__global__ void qkv_kernel(const _Float16* __restrict__ wq,   // [1536][512]
                           const float*    __restrict__ bias, // [1536]
                           const _Float16* __restrict__ xnT,  // [B][N][C]
                           _Float16* __restrict__ qT,
                           _Float16* __restrict__ kT,
                           _Float16* __restrict__ vC) {
  const int tiles_n = N_ / 32;           // 32
  const int tiles_m = (3 * C_) / 32;     // 48
  const int wid  = (blockIdx.x * blockDim.x + threadIdx.x) >> 5;
  const int lane = threadIdx.x & 31;
  const int per_b = tiles_m * tiles_n;
  const int b  = wid / per_b;
  const int r  = wid % per_b;
  const int tm = r / tiles_n;
  const int tn = r % tiles_n;

  const _Float16* wt0 = wq  + (size_t)tm * 32 * C_;
  const _Float16* wt1 = wt0 + 16 * C_;
  const _Float16* xt0 = xnT + ((size_t)b * N_ + tn * 32) * C_;
  const _Float16* xt1 = xt0 + 16 * C_;

  v8f a00 = {}, a01 = {}, a10 = {}, a11 = {};
  for (int k = 0; k < C_; k += 32) {
    const v16h fa0 = load_frag16x32(wt0 + k, C_);
    const v16h fa1 = load_frag16x32(wt1 + k, C_);
    const v16h fb0 = load_frag16x32(xt0 + k, C_);
    const v16h fb1 = load_frag16x32(xt1 + k, C_);
    a00 = wmma32(fa0, fb0, a00);
    a01 = wmma32(fa0, fb1, a01);
    a10 = wmma32(fa1, fb0, a10);
    a11 = wmma32(fa1, fb1, a11);
  }

  const int col = lane & 15;
  qkv_store_tile(a00, b, tm * 32 + 0,  tn * 32 + 0  + col, bias, qT, kT, vC);
  qkv_store_tile(a01, b, tm * 32 + 0,  tn * 32 + 16 + col, bias, qT, kT, vC);
  qkv_store_tile(a10, b, tm * 32 + 16, tn * 32 + 0  + col, bias, qT, kT, vC);
  qkv_store_tile(a11, b, tm * 32 + 16, tn * 32 + 16 + col, bias, qT, kT, vC);
}

// --------------------------------------------- online softmax for one row tile
__device__ __forceinline__ void online_softmax(v8f& s0, v8f& s1,
                                               float* m_i, float* l_i,
                                               float* scale) {
  #pragma unroll
  for (int i = 0; i < 8; ++i) {
    float mx = fmaxf(s0[i], s1[i]);
    mx = fmaxf(mx, __shfl_xor(mx, 1, 32));
    mx = fmaxf(mx, __shfl_xor(mx, 2, 32));
    mx = fmaxf(mx, __shfl_xor(mx, 4, 32));
    mx = fmaxf(mx, __shfl_xor(mx, 8, 32));
    const float mnew = fmaxf(m_i[i], mx);
    const float p0 = __expf(s0[i] - mnew);
    const float p1 = __expf(s1[i] - mnew);
    float rs = p0 + p1;
    rs += __shfl_xor(rs, 1, 32);
    rs += __shfl_xor(rs, 2, 32);
    rs += __shfl_xor(rs, 4, 32);
    rs += __shfl_xor(rs, 8, 32);
    const float corr = __expf(m_i[i] - mnew);
    l_i[i] = l_i[i] * corr + rs;
    m_i[i] = mnew;
    scale[i] = corr;
    s0[i] = p0; s1[i] = p1;
  }
}

// ---- flash attention: one wave per (b,h, 32-row slab = two 16-row WMMA tiles),
// sharing every K/V fragment between the two row tiles (halves L2 traffic).
__global__ void attn_kernel(const _Float16* __restrict__ qT,  // [B][H][N][64]
                            const _Float16* __restrict__ kT,  // [B][H][N][64]
                            const _Float16* __restrict__ vC,  // [B][H][64][N]
                            _Float16* __restrict__ oT) {      // [B][N][512]
  __shared__ _Float16 lds_p[8][2][16 * 32];       // per-wave P staging tiles
  const int wib  = threadIdx.x >> 5;
  const int lane = threadIdx.x & 31;
  const int gw = blockIdx.x * (blockDim.x >> 5) + wib;
  const int it = gw & 31;          // 32-row slab index
  const int h  = (gw >> 5) & 7;
  const int b  = gw >> 8;

  const _Float16* qb = qT + (((size_t)b * H_ + h) * N_ + it * 32) * CH_;
  const _Float16* kb = kT + ((size_t)b * H_ + h) * N_ * CH_;
  const _Float16* vb = vC + ((size_t)b * H_ + h) * CH_ * N_;

  const v16h aqA0 = load_frag16x32(qb, CH_);
  const v16h aqA1 = load_frag16x32(qb + 32, CH_);
  const v16h aqB0 = load_frag16x32(qb + 16 * CH_, CH_);
  const v16h aqB1 = load_frag16x32(qb + 16 * CH_ + 32, CH_);

  v8f accA0 = {}, accA1 = {}, accA2 = {}, accA3 = {};
  v8f accB0 = {}, accB1 = {}, accB2 = {}, accB3 = {};
  float mA[8], lA[8], mB[8], lB[8];
  #pragma unroll
  for (int i = 0; i < 8; ++i) {
    mA[i] = -3.0e30f; lA[i] = 0.f;
    mB[i] = -3.0e30f; lB[i] = 0.f;
  }

  _Float16* lpA = &lds_p[wib][0][0];
  _Float16* lpB = &lds_p[wib][1][0];
  const int off = (lane >> 4) << 3;
  const int col = lane & 15;

  for (int j = 0; j < N_; j += 32) {
    // shared K fragments for two 16-column score tiles (K = 64 head dim)
    const v16h kf0 = load_frag16x32(kb + (size_t)j * CH_,             CH_);
    const v16h kf1 = load_frag16x32(kb + (size_t)j * CH_ + 32,        CH_);
    const v16h kf2 = load_frag16x32(kb + (size_t)(j + 16) * CH_,      CH_);
    const v16h kf3 = load_frag16x32(kb + (size_t)(j + 16) * CH_ + 32, CH_);

    v8f sA0 = {}, sA1 = {}, sB0 = {}, sB1 = {};
    sA0 = wmma32(aqA0, kf0, sA0);  sA0 = wmma32(aqA1, kf1, sA0);
    sA1 = wmma32(aqA0, kf2, sA1);  sA1 = wmma32(aqA1, kf3, sA1);
    sB0 = wmma32(aqB0, kf0, sB0);  sB0 = wmma32(aqB1, kf1, sB0);
    sB1 = wmma32(aqB0, kf2, sB1);  sB1 = wmma32(aqB1, kf3, sB1);

    float scA[8], scB[8];
    online_softmax(sA0, sA1, mA, lA, scA);
    online_softmax(sB0, sB1, mB, lB, scB);
    #pragma unroll
    for (int i = 0; i < 8; ++i) {
      accA0[i] *= scA[i]; accA1[i] *= scA[i];
      accA2[i] *= scA[i]; accA3[i] *= scA[i];
      accB0[i] *= scB[i]; accB1[i] *= scB[i];
      accB2[i] *= scB[i]; accB3[i] *= scB[i];
    }

    // transpose P (C/D layout -> A layout) through LDS; wave-private tiles,
    // LDS ops are in-order within a wave, explicit dscnt wait before reload
    #pragma unroll
    for (int i = 0; i < 8; ++i) {
      lpA[(off + i) * 32 + col]      = (_Float16)sA0[i];
      lpA[(off + i) * 32 + 16 + col] = (_Float16)sA1[i];
      lpB[(off + i) * 32 + col]      = (_Float16)sB0[i];
      lpB[(off + i) * 32 + 16 + col] = (_Float16)sB1[i];
    }
    asm volatile("s_wait_dscnt 0x0" ::: "memory");
    const v16h apA = load_frag16x32(lpA, 32);
    const v16h apB = load_frag16x32(lpB, 32);

    // shared V fragments feed both row tiles: O += P @ V
    const v16h vf0 = load_frag16x32(vb + 0 * 16 * N_ + j, N_);
    const v16h vf1 = load_frag16x32(vb + 1 * 16 * N_ + j, N_);
    const v16h vf2 = load_frag16x32(vb + 2 * 16 * N_ + j, N_);
    const v16h vf3 = load_frag16x32(vb + 3 * 16 * N_ + j, N_);
    accA0 = wmma32(apA, vf0, accA0);  accB0 = wmma32(apB, vf0, accB0);
    accA1 = wmma32(apA, vf1, accA1);  accB1 = wmma32(apB, vf1, accB1);
    accA2 = wmma32(apA, vf2, accA2);  accB2 = wmma32(apB, vf2, accB2);
    accA3 = wmma32(apA, vf3, accA3);  accB3 = wmma32(apB, vf3, accB3);
  }

  // finalize: divide each row by its softmax sum, write oT[b][n][h*64 + c]
  #pragma unroll
  for (int i = 0; i < 8; ++i) {
    const float invA = 1.0f / lA[i];
    const int nA = it * 32 + off + i;
    _Float16* dA = oT + ((size_t)b * N_ + nA) * C_ + h * CH_;
    dA[0 * 16 + col] = (_Float16)(accA0[i] * invA);
    dA[1 * 16 + col] = (_Float16)(accA1[i] * invA);
    dA[2 * 16 + col] = (_Float16)(accA2[i] * invA);
    dA[3 * 16 + col] = (_Float16)(accA3[i] * invA);

    const float invB = 1.0f / lB[i];
    const int nB = it * 32 + 16 + off + i;
    _Float16* dB = oT + ((size_t)b * N_ + nB) * C_ + h * CH_;
    dB[0 * 16 + col] = (_Float16)(accB0[i] * invB);
    dB[1 * 16 + col] = (_Float16)(accB1[i] * invB);
    dB[2 * 16 + col] = (_Float16)(accB2[i] * invB);
    dB[3 * 16 + col] = (_Float16)(accB3[i] * invB);
  }
}

// ------- proj GEMM + bias + residual, 32x32 register-blocked per wave
__global__ void proj_kernel(const _Float16* __restrict__ wp,  // [512][512]
                            const float*    __restrict__ pb,  // [512]
                            const _Float16* __restrict__ oT,  // [B][N][C]
                            const float*    __restrict__ x,   // [B][C][N]
                            float* __restrict__ out) {        // [B][C][N]
  const int tiles_n = N_ / 32;   // 32
  const int tiles_m = C_ / 32;   // 16
  const int wid  = (blockIdx.x * blockDim.x + threadIdx.x) >> 5;
  const int lane = threadIdx.x & 31;
  const int per_b = tiles_m * tiles_n;
  const int b  = wid / per_b;
  const int r  = wid % per_b;
  const int tm = r / tiles_n;
  const int tn = r % tiles_n;

  const _Float16* wt0 = wp + (size_t)tm * 32 * C_;
  const _Float16* wt1 = wt0 + 16 * C_;
  const _Float16* ot0 = oT + ((size_t)b * N_ + tn * 32) * C_;
  const _Float16* ot1 = ot0 + 16 * C_;

  v8f a00 = {}, a01 = {}, a10 = {}, a11 = {};
  for (int k = 0; k < C_; k += 32) {
    const v16h fa0 = load_frag16x32(wt0 + k, C_);
    const v16h fa1 = load_frag16x32(wt1 + k, C_);
    const v16h fb0 = load_frag16x32(ot0 + k, C_);
    const v16h fb1 = load_frag16x32(ot1 + k, C_);
    a00 = wmma32(fa0, fb0, a00);
    a01 = wmma32(fa0, fb1, a01);
    a10 = wmma32(fa1, fb0, a10);
    a11 = wmma32(fa1, fb1, a11);
  }

  const int off = (lane >> 4) << 3;
  const int col = lane & 15;
  v8f accs[2][2] = {{a00, a01}, {a10, a11}};
  #pragma unroll
  for (int mt = 0; mt < 2; ++mt) {
    const float* bp = pb + tm * 32 + mt * 16 + off;
    float4 b0 = *(const float4*)bp;
    float4 b1 = *(const float4*)(bp + 4);
    float bv[8] = {b0.x, b0.y, b0.z, b0.w, b1.x, b1.y, b1.z, b1.w};
    #pragma unroll
    for (int nt = 0; nt < 2; ++nt) {
      const int n = tn * 32 + nt * 16 + col;
      #pragma unroll
      for (int i = 0; i < 8; ++i) {
        const int o = tm * 32 + mt * 16 + off + i;
        const size_t idx = ((size_t)b * C_ + o) * N_ + n;
        out[idx] = accs[mt][nt][i] + bv[i] + x[idx];
      }
    }
  }
}

// ------------------------------------------------------------------ launch
extern "C" void kernel_launch(void* const* d_in, const int* in_sizes, int n_in,
                              void* d_out, int out_size, void* d_ws, size_t ws_size,
                              hipStream_t stream) {
  const float* x      = (const float*)d_in[0];
  const float* norm_w = (const float*)d_in[1];
  const float* norm_b = (const float*)d_in[2];
  const float* qkv_w  = (const float*)d_in[3];
  const float* qkv_b  = (const float*)d_in[4];
  const float* proj_w = (const float*)d_in[5];
  const float* proj_b = (const float*)d_in[6];
  float* out = (float*)d_out;

  char* ws = (char*)d_ws;
  _Float16* wq_h = (_Float16*)ws;  ws += (size_t)3 * C_ * C_ * 2;       // 1.5 MB
  _Float16* wp_h = (_Float16*)ws;  ws += (size_t)C_ * C_ * 2;           // 0.5 MB
  _Float16* xnT  = (_Float16*)ws;  ws += (size_t)B_ * N_ * C_ * 2;      // 16 MB
  _Float16* qT   = (_Float16*)ws;  ws += (size_t)B_ * H_ * N_ * CH_ * 2;
  _Float16* kT   = (_Float16*)ws;  ws += (size_t)B_ * H_ * N_ * CH_ * 2;
  _Float16* vC   = (_Float16*)ws;  ws += (size_t)B_ * H_ * CH_ * N_ * 2;
  _Float16* oT   = (_Float16*)ws;  ws += (size_t)B_ * N_ * C_ * 2;

  // 1) weights to f16
  cvt_f16_kernel<<<(3 * C_ * C_) / 256, 256, 0, stream>>>(qkv_w, wq_h, 3 * C_ * C_);
  cvt_f16_kernel<<<(C_ * C_) / 256, 256, 0, stream>>>(proj_w, wp_h, C_ * C_);

  // 2) groupnorm (one block per (batch, group))
  gn_kernel<<<B_ * G_, 256, 0, stream>>>(x, norm_w, norm_b, xnT);

  // 3) QKV gemm: 16 * 48 * 32 = 24576 waves, 8 waves/block
  qkv_kernel<<<(B_ * 48 * 32) / 8, 256, 0, stream>>>(wq_h, qkv_b, xnT, qT, kT, vC);

  // 4) flash attention: 16 * 8 * 32 = 4096 waves
  attn_kernel<<<(B_ * H_ * 32) / 8, 256, 0, stream>>>(qT, kT, vC, oT);

  // 5) proj + residual: 16 * 16 * 32 = 8192 waves
  proj_kernel<<<(B_ * 16 * 32) / 8, 256, 0, stream>>>(wp_h, proj_b, oT, x, out);
}
